// DifferentiableLayer_11218454577800
// MI455X (gfx1250) — matrix operand order repaired and checked by reading, not compiled, not run on hardware
//
#include <hip/hip_runtime.h>

typedef __attribute__((ext_vector_type(16))) _Float16 v16h;
typedef __attribute__((ext_vector_type(8)))  _Float16 v8h;
typedef __attribute__((ext_vector_type(8)))  float    v8f;

#define B_DIM 4096
#define I_DIM 512
#define O_DIM 512
#define K_OPS 6
#define KD    (I_DIM * K_OPS)   // 3072 flattened contraction dim

union H2U { _Float16 h[2]; unsigned u; };
union V16 { v16h v; v8h h[2]; };

// ---------------------------------------------------------------------------
// Prep 1: op bank.  Xop[b][i*6+k] = op_k(x[b,i]) in f16, row-major [B, KD].
// x is read exactly once -> non-temporal loads; Xop is re-read by the GEMM
// -> default (RT) stores so it stays resident in the 192MB L2.
// ---------------------------------------------------------------------------
__global__ __launch_bounds__(256) void build_xop(const float* __restrict__ x,
                                                 _Float16* __restrict__ Xop) {
    int idx = blockIdx.x * blockDim.x + threadIdx.x;   // over B*I
    if (idx >= B_DIM * I_DIM) return;
    float v = __builtin_nontemporal_load(x + idx);
    float ops[6];
    ops[0] = v;                       // identity
    ops[1] = __sinf(v);               // sin
    ops[2] = __cosf(v);               // cos
    ops[3] = tanhf(v);                // tanh
    ops[4] = v * v;                   // square
    ops[5] = v > 0.f ? v : 0.f;       // relu
    unsigned* out = (unsigned*)(Xop + (size_t)idx * K_OPS);  // 4B aligned (12B stride)
#pragma unroll
    for (int j = 0; j < 3; ++j) {
        H2U p;
        p.h[0] = (_Float16)ops[2 * j];
        p.h[1] = (_Float16)ops[2 * j + 1];
        out[j] = p.u;
    }
}

// ---------------------------------------------------------------------------
// Prep 2: hard Gumbel-softmax selection -> one-hot f16 matrix.
// Pcol stored [O][KD] row-major (B-matrix transposed: column o contiguous in
// K) so each WMMA B-fragment lane reads two runs of 8 contiguous f16.
// weights/gumbel read once -> non-temporal loads.
// ---------------------------------------------------------------------------
__global__ __launch_bounds__(256) void build_p(const float* __restrict__ w,
                                               const float* __restrict__ g,
                                               _Float16* __restrict__ Pcol) {
    int idx = blockIdx.x * blockDim.x + threadIdx.x;   // over O*I
    if (idx >= O_DIM * I_DIM) return;
    const float* wp = w + (size_t)idx * K_OPS;
    const float* gp = g + (size_t)idx * K_OPS;
    float best = __builtin_nontemporal_load(wp) + __builtin_nontemporal_load(gp);
    int bi = 0;
#pragma unroll
    for (int k = 1; k < K_OPS; ++k) {
        float s = __builtin_nontemporal_load(wp + k) + __builtin_nontemporal_load(gp + k);
        if (s > best) { best = s; bi = k; }   // first-max tie-break like jnp.argmax
    }
    int o = idx / I_DIM;
    int i = idx - o * I_DIM;
    unsigned* out = (unsigned*)(Pcol + (size_t)o * KD + (size_t)i * K_OPS);
#pragma unroll
    for (int j = 0; j < 3; ++j) {
        H2U p;
        p.h[0] = (_Float16)(bi == 2 * j     ? 1.f : 0.f);
        p.h[1] = (_Float16)(bi == 2 * j + 1 ? 1.f : 0.f);
        out[j] = p.u;
    }
}

// ---------------------------------------------------------------------------
// GEMM: out[4096,512] = Xop[4096,3072] x Pcol^T  via v_wmma_f32_16x16x32_f16.
// Block = 256 threads = 8 waves (wave32). Wave tile 32x64 (2x4 WMMA tiles),
// block tile 128x128 (4M x 2N waves). K-loop: 3072/32 = 96 steps, 8 WMMA per
// step per wave. Intra-block fragment redundancy (Bx4, Ax2) is absorbed by
// the WGP$ since all 8 waves walk the same K-step. Output is written with
// non-temporal stores (never re-read) to preserve the L2-resident A/B set.
// ---------------------------------------------------------------------------
__global__ __launch_bounds__(256) void gemm_wmma(const _Float16* __restrict__ Xop,
                                                 const _Float16* __restrict__ Pcol,
                                                 float* __restrict__ C) {
    const int lane  = threadIdx.x & 31;
    const int wave  = threadIdx.x >> 5;   // 0..7
    const int waveM = wave >> 1;          // 0..3 -> M sub-tile
    const int waveN = wave & 1;           // 0..1 -> N sub-tile

    const int tileM = blockIdx.x * 128 + waveM * 32;
    const int tileN = blockIdx.y * 128 + waveN * 64;

    const int half = lane >> 4;           // 0: lanes 0-15, 1: lanes 16-31
    const int l15  = lane & 15;

    // A fragment (16x32 f16, ISA 7.12.2): lane half 0 -> K 0..7 & 16..23,
    // lane half 1 -> K 8..15 & 24..31; row M = lane&15.
    const _Float16* aBase0 = Xop + (size_t)(tileM + l15) * KD + (half ? 8 : 0);
    const _Float16* aBase1 = aBase0 + (size_t)16 * KD;

    // B fragment (32x16 f16): lane half 0 -> K 0..7 & 16..23, half 1 -> K 8..15 & 24..31;
    // column N = lane&15. Pcol row o == column N, contiguous in K.
    const _Float16* bBase = Pcol + (size_t)(tileN + l15) * KD + (half ? 8 : 0);

    v8f acc[2][4];
#pragma unroll
    for (int mi = 0; mi < 2; ++mi)
#pragma unroll
        for (int ni = 0; ni < 4; ++ni)
            acc[mi][ni] = (v8f)(0.f);

#pragma unroll 2
    for (int kt = 0; kt < KD; kt += 32) {
        v16h a[2], b[4];
#pragma unroll
        for (int mi = 0; mi < 2; ++mi) {
            const _Float16* p = (mi ? aBase1 : aBase0) + kt;
            V16 t;
            t.h[0] = *(const v8h*)p;          // K kb..kb+7
            t.h[1] = *(const v8h*)(p + 16);   // K kb+16..kb+23
            a[mi] = t.v;
        }
#pragma unroll
        for (int ni = 0; ni < 4; ++ni) {
            const _Float16* p = bBase + (size_t)ni * 16 * KD + kt;
            V16 t;
            t.h[0] = *(const v8h*)p;          // K kb..kb+7
            t.h[1] = *(const v8h*)(p + 16);   // K kb+16..kb+23
            b[ni] = t.v;
        }
#pragma unroll
        for (int mi = 0; mi < 2; ++mi)
#pragma unroll
            for (int ni = 0; ni < 4; ++ni)
                acc[mi][ni] = __builtin_amdgcn_wmma_f32_16x16x32_f16(
                    /*neg_a=*/false, a[mi], /*neg_b=*/false, b[ni],
                    /*c_mod=*/(short)0, acc[mi][ni],
                    /*reuse_a=*/false, /*reuse_b=*/false);
    }

    // D layout: lanes 0-15: VGPR r -> (M=r, N=lane); lanes 16-31: (M=8+r, N=lane-16)
    const int m0 = half * 8;
#pragma unroll
    for (int mi = 0; mi < 2; ++mi)
#pragma unroll
        for (int ni = 0; ni < 4; ++ni) {
            float* cp = C + (size_t)(tileM + mi * 16 + m0) * O_DIM
                          + (size_t)(tileN + ni * 16 + l15);
#pragma unroll
            for (int r = 0; r < 8; ++r)
                __builtin_nontemporal_store(acc[mi][ni][r], cp + (size_t)r * O_DIM);
        }
}

// ---------------------------------------------------------------------------
extern "C" void kernel_launch(void* const* d_in, const int* in_sizes, int n_in,
                              void* d_out, int out_size, void* d_ws, size_t ws_size,
                              hipStream_t stream) {
    const float* x = (const float*)d_in[0];   // [4096, 512]
    const float* w = (const float*)d_in[1];   // [512, 512, 6]
    const float* g = (const float*)d_in[2];   // [512, 512, 6]
    float* out = (float*)d_out;               // [4096, 512]

    _Float16* Xop  = (_Float16*)d_ws;                         // 24 MB
    _Float16* Pcol = Xop + (size_t)B_DIM * KD;                // +3 MB

    build_xop<<<(B_DIM * I_DIM + 255) / 256, 256, 0, stream>>>(x, Xop);
    build_p<<<(O_DIM * I_DIM + 255) / 256, 256, 0, stream>>>(w, g, Pcol);

    dim3 grid(B_DIM / 128, O_DIM / 128);      // 32 x 4 blocks, 8 waves each
    gemm_wmma<<<grid, 256, 0, stream>>>(Xop, Pcol, out);
}
